// SpatialLRIGNN_10565619548640
// MI455X (gfx1250) — matrix operand order repaired and checked by reading, not compiled, not run on hardware
//
#include <hip/hip_runtime.h>
#include <hip/hip_bf16.h>

#define N_NODES 50000
#define N_EDGES 800000
#define IN_DIM  384
#define HID     128
#define OUT_DIM 32
#define N_GRAPHS 8
#define NEG_SLOPE 0.2f

typedef __attribute__((ext_vector_type(16))) _Float16 v16h;
typedef __attribute__((ext_vector_type(8)))  _Float16 v8h;
typedef __attribute__((ext_vector_type(8)))  float    v8f;
typedef __attribute__((ext_vector_type(4)))  float    v4f;

// ---------------- utility kernels ----------------

__global__ void k_f32_to_f16(const float* __restrict__ in, _Float16* __restrict__ out, int n) {
    int i = blockIdx.x * blockDim.x + threadIdx.x;
    if (i < n) out[i] = (_Float16)in[i];
}

__global__ void k_relu_f32_to_f16(const float* __restrict__ in, _Float16* __restrict__ out, int n) {
    int i = blockIdx.x * blockDim.x + threadIdx.x;
    if (i < n) {
        float v = in[i];
        out[i] = (_Float16)(v > 0.f ? v : 0.f);
    }
}

__global__ void k_fill(float* __restrict__ p, float v, int n) {
    int i = blockIdx.x * blockDim.x + threadIdx.x;
    if (i < n) p[i] = v;
}

__global__ void k_init_bias(float* __restrict__ out, const float* __restrict__ bias, int n, int stride) {
    int i = blockIdx.x * blockDim.x + threadIdx.x;
    if (i < n) out[i] = bias[i % stride];
}

// Pack row-major f32 W[K][N] into WMMA B-fragment order (f16):
// Bp[(((nt*Kt + kt)*32 + lane)*16 + j)] = W[kt*32 + j + 16*(lane>>4)][nt*16 + (lane&15)]
__global__ void k_pack_b(const float* __restrict__ W, _Float16* __restrict__ Bp, int K, int N) {
    int t = blockIdx.x * blockDim.x + threadIdx.x;
    int total = K * N;
    if (t >= total) return;
    int j    = t & 15;
    int lane = (t >> 4) & 31;
    int Kt   = K >> 5;
    int kt   = (t >> 9) % Kt;
    int nt   = t / (Kt << 9);
    int hi   = lane >> 4;
    int n    = nt * 16 + (lane & 15);
    int k    = kt * 32 + j + 16 * hi;
    Bp[t] = (_Float16)W[(size_t)k * N + n];
}

// ---------------- WMMA GEMM ----------------
// C[M x N] = A[M x K] * B[K x N], A f16 row-major, B pre-packed fragments, C f32 row-major.
// grid.x = M/16 (M-tiles), block = 128 threads = 4 waves; wave w owns N-tiles [w*NTW, (w+1)*NTW)
template<int NTW>
__global__ void k_gemm_wmma(const _Float16* __restrict__ A, const _Float16* __restrict__ Bp,
                            float* __restrict__ C, int K) {
    constexpr int NTILES = 4 * NTW;
    constexpr int N = 16 * NTILES;
    const int wave = threadIdx.x >> 5;
    const int lane = threadIdx.x & 31;
    const int hi   = lane >> 4;
    const int mt   = blockIdx.x;
    const int Kt   = K >> 5;
    const int m    = mt * 16 + (lane & 15);

    v8f acc[NTW];
#pragma unroll
    for (int i = 0; i < NTW; ++i) acc[i] = (v8f){0.f,0.f,0.f,0.f,0.f,0.f,0.f,0.f};

    const _Float16* arow = A + (size_t)m * K + hi * 16;
    for (int kt = 0; kt < Kt; ++kt) {
        // A fragment: lane<16 holds K {0..15}, lane>=16 holds K {16..31}
        v8h a0 = *(const v8h*)(arow + kt * 32);
        v8h a1 = *(const v8h*)(arow + kt * 32 + 8);
        v16h af;
#pragma unroll
        for (int j = 0; j < 8; ++j) { af[j] = a0[j]; af[j + 8] = a1[j]; }
#pragma unroll
        for (int i = 0; i < NTW; ++i) {
            int nt = wave * NTW + i;
            v16h bf = *(const v16h*)(Bp + (((size_t)nt * Kt + kt) * 32 + lane) * 16);
            acc[i] = __builtin_amdgcn_wmma_f32_16x16x32_f16(
                false, af, false, bf, (short)0, acc[i], false, false);
        }
    }
    // D layout: reg r -> row (r + 8*hi), col (lane&15) within 16x16 tile
#pragma unroll
    for (int i = 0; i < NTW; ++i) {
        int nt = wave * NTW + i;
        int n  = nt * 16 + (lane & 15);
#pragma unroll
        for (int r = 0; r < 8; ++r) {
            int mm = mt * 16 + r + 8 * hi;
            C[(size_t)mm * N + n] = acc[i][r];
        }
    }
}

// ---------------- attention scores ----------------
// one wave per (node, head): a_s[n,h] = sum_c h[n,h,c]*att_s[h,c] (HID=128, 4 floats/lane)
__global__ void k_att_scores(const float* __restrict__ h, const float* __restrict__ att_s,
                             const float* __restrict__ att_d, float* __restrict__ a_s,
                             float* __restrict__ a_d, int heads) {
    int lane = threadIdx.x & 31;
    int wid  = blockIdx.x * (blockDim.x >> 5) + (threadIdx.x >> 5);
    if (wid >= N_NODES * heads) return;
    int n  = wid / heads;
    int hh = wid - n * heads;
    const float* hp = h + (size_t)n * heads * HID + hh * HID + lane * 4;
    const float* sp = att_s + hh * HID + lane * 4;
    const float* dp = att_d + hh * HID + lane * 4;
    v4f hv = *(const v4f*)hp;
    v4f sv = *(const v4f*)sp;
    v4f dv = *(const v4f*)dp;
    float ss = hv[0]*sv[0] + hv[1]*sv[1] + hv[2]*sv[2] + hv[3]*sv[3];
    float sd = hv[0]*dv[0] + hv[1]*dv[1] + hv[2]*dv[2] + hv[3]*dv[3];
#pragma unroll
    for (int off = 16; off > 0; off >>= 1) {
        ss += __shfl_xor(ss, off, 32);
        sd += __shfl_xor(sd, off, 32);
    }
    if (lane == 0) { a_s[wid] = ss; a_d[wid] = sd; }
}

// ---------------- edge softmax phases ----------------
__device__ inline void atomicMaxF(float* addr, float val) {
    unsigned int* ua = (unsigned int*)addr;
    unsigned int old = *ua;
    while (__uint_as_float(old) < val) {
        unsigned int prev = atomicCAS(ua, old, __float_as_uint(val));
        if (prev == old) break;
        old = prev;
    }
}

__global__ void k_edge_logits(const int* __restrict__ src, const int* __restrict__ dst,
                              const float* __restrict__ a_s, const float* __restrict__ a_d,
                              float* __restrict__ elog, float* __restrict__ mmax, int heads) {
    int idx = blockIdx.x * blockDim.x + threadIdx.x;
    if (idx >= N_EDGES * heads) return;
    int e  = idx / heads;
    int hh = idx - e * heads;
    int s = src[e], d = dst[e];
    float v = a_s[s * heads + hh] + a_d[d * heads + hh];
    v = v > 0.f ? v : NEG_SLOPE * v;
    elog[idx] = v;
    atomicMaxF(&mmax[d * heads + hh], v);
}

__global__ void k_edge_exp(const int* __restrict__ dst, float* __restrict__ ex,
                           const float* __restrict__ mmax, float* __restrict__ den, int heads) {
    int idx = blockIdx.x * blockDim.x + threadIdx.x;
    if (idx >= N_EDGES * heads) return;
    int e  = idx / heads;
    int hh = idx - e * heads;
    int d = dst[e];
    float v = __expf(ex[idx] - mmax[d * heads + hh]);
    ex[idx] = v;
    atomicAdd(&den[d * heads + hh], v);
}

// wave per (edge, head); lane handles 4 channels: out[dst] += h[src] * coef
__global__ void k_edge_agg(const int* __restrict__ src, const int* __restrict__ dst,
                           const float* __restrict__ ex, const float* __restrict__ den,
                           const float* __restrict__ ew, const float* __restrict__ h,
                           float* __restrict__ out, int heads) {
    int lane = threadIdx.x & 31;
    int wid  = blockIdx.x * (blockDim.x >> 5) + (threadIdx.x >> 5);
    if (wid >= N_EDGES * heads) return;
    int e  = wid / heads;
    int hh = wid - e * heads;
    int s = src[e], d = dst[e];
    float coef = ex[wid] / (den[d * heads + hh] + 1e-16f) * ew[e];
    const float* hp = h + (size_t)s * heads * HID + hh * HID + lane * 4;
    float* op = out + (size_t)d * heads * HID + hh * HID + lane * 4;
    v4f hv = *(const v4f*)hp;
    atomicAdd(op + 0, hv[0] * coef);
    atomicAdd(op + 1, hv[1] * coef);
    atomicAdd(op + 2, hv[2] * coef);
    atomicAdd(op + 3, hv[3] * coef);
}

// ---------------- pooling + decoder ----------------
__global__ void k_pool(const float* __restrict__ h, const int* __restrict__ batch,
                       float* __restrict__ sums, float* __restrict__ counts) {
    int lane = threadIdx.x & 31;
    int wid  = blockIdx.x * (blockDim.x >> 5) + (threadIdx.x >> 5);
    if (wid >= N_NODES) return;
    int g = batch[wid];
    v4f hv = *(const v4f*)(h + (size_t)wid * HID + lane * 4);
    float* sp = sums + g * HID + lane * 4;
    atomicAdd(sp + 0, hv[0]);
    atomicAdd(sp + 1, hv[1]);
    atomicAdd(sp + 2, hv[2]);
    atomicAdd(sp + 3, hv[3]);
    if (lane == 0) atomicAdd(&counts[g], 1.0f);
}

__global__ void k_decoder(const float* __restrict__ sums, const float* __restrict__ counts,
                          const float* __restrict__ Wd1, const float* __restrict__ bd1,
                          const float* __restrict__ Wd2, const float* __restrict__ bd2,
                          float* __restrict__ out) {
    __shared__ float p[N_GRAPHS * HID];
    __shared__ float z[N_GRAPHS * HID];
    int tid = threadIdx.x;
    for (int i = tid; i < N_GRAPHS * HID; i += blockDim.x) {
        int g = i >> 7;
        p[i] = sums[i] / fmaxf(counts[g], 1.0f);
    }
    __syncthreads();
    for (int i = tid; i < N_GRAPHS * HID; i += blockDim.x) {
        int g = i >> 7, o = i & 127;
        float acc = bd1[o];
        for (int k = 0; k < HID; ++k) acc += p[g * HID + k] * Wd1[k * HID + o];
        z[i] = acc > 0.f ? acc : 0.f;
    }
    __syncthreads();
    for (int i = tid; i < N_GRAPHS * OUT_DIM; i += blockDim.x) {
        int g = i >> 5, o = i & 31;
        float acc = bd2[o];
        for (int k = 0; k < HID; ++k) acc += z[g * HID + k] * Wd2[k * OUT_DIM + o];
        out[i] = acc;
    }
}

// ---------------- launch ----------------
extern "C" void kernel_launch(void* const* d_in, const int* in_sizes, int n_in,
                              void* d_out, int out_size, void* d_ws, size_t ws_size,
                              hipStream_t stream) {
    (void)in_sizes; (void)n_in; (void)out_size; (void)ws_size;
    const float* x        = (const float*)d_in[0];
    const int*   src      = (const int*)d_in[1];
    const int*   dst      = ((const int*)d_in[1]) + N_EDGES;
    const float* ew       = (const float*)d_in[2];
    const int*   batch    = (const int*)d_in[3];
    const float* W1       = (const float*)d_in[4];
    const float* att_src1 = (const float*)d_in[5];
    const float* att_dst1 = (const float*)d_in[6];
    const float* b1       = (const float*)d_in[7];
    const float* W2       = (const float*)d_in[8];
    const float* att_src2 = (const float*)d_in[9];
    const float* att_dst2 = (const float*)d_in[10];
    const float* b2       = (const float*)d_in[11];
    const float* Wd1      = (const float*)d_in[12];
    const float* bd1      = (const float*)d_in[13];
    const float* Wd2      = (const float*)d_in[14];
    const float* bd2      = (const float*)d_in[15];

    // workspace layout (256B aligned slabs, with reuse)
    char* base = (char*)d_ws;
    size_t off = 0;
    auto take = [&](size_t bytes) -> char* {
        char* p = base + off;
        off += (bytes + 255) & ~(size_t)255;
        return p;
    };
    _Float16* Wp1   = (_Float16*)take((size_t)IN_DIM * 256 * 2);     // packed W1 frags
    _Float16* Wp2   = (_Float16*)take((size_t)256 * HID * 2);        // packed W2 frags
    char*     regC  = take((size_t)N_NODES * IN_DIM * 2);            // xh, later h1act(f16)
    float*    h1lin = (float*)take((size_t)N_NODES * 256 * 4);       // later: h2lin + out2
    float*    out1  = (float*)take((size_t)N_NODES * 256 * 4);
    float*    a_s   = (float*)take((size_t)N_NODES * 2 * 4);
    float*    a_d   = (float*)take((size_t)N_NODES * 2 * 4);
    float*    mbuf  = (float*)take((size_t)N_NODES * 2 * 4);
    float*    dbuf  = (float*)take((size_t)N_NODES * 2 * 4);
    float*    exbuf = (float*)take((size_t)N_EDGES * 2 * 4);
    float*    sums  = (float*)take((size_t)N_GRAPHS * HID * 4);
    float*    counts= (float*)take((size_t)N_GRAPHS * 4);
    _Float16* xh    = (_Float16*)regC;
    _Float16* h1act = (_Float16*)regC;                               // reuse (25.6MB <= 38.4MB)
    float*    h2lin = h1lin;                                         // reuse first half of D
    float*    out2  = (float*)((char*)h1lin + (size_t)N_NODES * HID * 4); // second half of D

    const int B = 256;
    auto blocks = [](long long n, int b) { return (unsigned)((n + b - 1) / b); };

    // ---- layer 1 ----
    k_pack_b<<<blocks((long long)IN_DIM * 256, B), B, 0, stream>>>(W1, Wp1, IN_DIM, 256);
    k_pack_b<<<blocks((long long)256 * HID, B), B, 0, stream>>>(W2, Wp2, 256, HID);
    k_f32_to_f16<<<blocks((long long)N_NODES * IN_DIM, B), B, 0, stream>>>(x, xh, N_NODES * IN_DIM);
    k_gemm_wmma<4><<<N_NODES / 16, 128, 0, stream>>>(xh, Wp1, h1lin, IN_DIM);

    k_att_scores<<<blocks((long long)N_NODES * 2 * 32, B), B, 0, stream>>>(h1lin, att_src1, att_dst1, a_s, a_d, 2);
    k_fill<<<blocks((long long)N_NODES * 2, B), B, 0, stream>>>(mbuf, -1e30f, N_NODES * 2);
    k_fill<<<blocks((long long)N_NODES * 2, B), B, 0, stream>>>(dbuf, 0.f, N_NODES * 2);
    k_init_bias<<<blocks((long long)N_NODES * 256, B), B, 0, stream>>>(out1, b1, N_NODES * 256, 256);
    k_edge_logits<<<blocks((long long)N_EDGES * 2, B), B, 0, stream>>>(src, dst, a_s, a_d, exbuf, mbuf, 2);
    k_edge_exp<<<blocks((long long)N_EDGES * 2, B), B, 0, stream>>>(dst, exbuf, mbuf, dbuf, 2);
    k_edge_agg<<<blocks((long long)N_EDGES * 2 * 32, B), B, 0, stream>>>(src, dst, exbuf, dbuf, ew, h1lin, out1, 2);
    k_relu_f32_to_f16<<<blocks((long long)N_NODES * 256, B), B, 0, stream>>>(out1, h1act, N_NODES * 256);

    // ---- layer 2 ----
    k_gemm_wmma<2><<<N_NODES / 16, 128, 0, stream>>>(h1act, Wp2, h2lin, 256);
    k_att_scores<<<blocks((long long)N_NODES * 32, B), B, 0, stream>>>(h2lin, att_src2, att_dst2, a_s, a_d, 1);
    k_fill<<<blocks((long long)N_NODES, B), B, 0, stream>>>(mbuf, -1e30f, N_NODES);
    k_fill<<<blocks((long long)N_NODES, B), B, 0, stream>>>(dbuf, 0.f, N_NODES);
    k_init_bias<<<blocks((long long)N_NODES * HID, B), B, 0, stream>>>(out2, b2, N_NODES * HID, HID);
    k_edge_logits<<<blocks((long long)N_EDGES, B), B, 0, stream>>>(src, dst, a_s, a_d, exbuf, mbuf, 1);
    k_edge_exp<<<blocks((long long)N_EDGES, B), B, 0, stream>>>(dst, exbuf, mbuf, dbuf, 1);
    k_edge_agg<<<blocks((long long)N_EDGES * 32, B), B, 0, stream>>>(src, dst, exbuf, dbuf, ew, h2lin, out2, 1);

    // ---- pool + decoder ----
    k_fill<<<blocks((long long)N_GRAPHS * HID, B), B, 0, stream>>>(sums, 0.f, N_GRAPHS * HID);
    k_fill<<<1, B, 0, stream>>>(counts, 0.f, N_GRAPHS);
    k_pool<<<blocks((long long)N_NODES * 32, B), B, 0, stream>>>(out2, batch, sums, counts);
    k_decoder<<<1, 256, 0, stream>>>(sums, counts, Wd1, bd1, Wd2, bd2, (float*)d_out);
}
